// MambaModel_32839319945567
// MI455X (gfx1250) — matrix operand order repaired and checked by reading, not compiled, not run on hardware
//
#include <hip/hip_runtime.h>
#include <stdint.h>

// ---------------- model dims ----------------
#define LAYERS   4
#define D_MODEL  768
#define D_STATE  16
#define D_CONV   4
#define D_INNER  1536
#define DT_RANK  48
#define B_SZ     4
#define SEQ      4096
#define MROWS    (B_SZ * SEQ)      // 16384
#define NCHUNK   16
#define CHUNK    (SEQ / NCHUNK)    // 256

#define USE_ASYNC 1
#define LPITCH   72                // LDS row pitch in bf16 (64 + 8 pad), 144B (16B-aligned)

typedef __bf16 bf16_t;
typedef __attribute__((ext_vector_type(16))) __bf16 v16bf;
typedef __attribute__((ext_vector_type(8)))  float  v8f;
typedef __attribute__((ext_vector_type(4)))  unsigned int u32x4;

union Frag { v16bf v; u32x4 q[2]; };

enum { EPI_NONE = 0, EPI_BIAS = 1, EPI_BIAS_RELU = 2, EPI_BIAS_SOFTPLUS = 3 };

__device__ __forceinline__ void wait_async0()
{
#if USE_ASYNC
#if __has_builtin(__builtin_amdgcn_s_wait_asynccnt)
    __builtin_amdgcn_s_wait_asynccnt(0);
#else
    asm volatile("s_wait_asynccnt 0x0" ::: "memory");
#endif
#endif
}

// ---------------------------------------------------------------------------
// bf16 WMMA GEMM with pre-packed operands:
//   A  : bf16 [M x KP]      (row-major, KP multiple of 64, zero-padded)
//   Bt : bf16 [Npad x KP]   (B transposed, Npad multiple of 128, zero-padded)
//   C  : f32 or bf16 [M x ldc]
// Block tile 128x128, 8 waves (2 M x 4 N), wave tile 64x32.
// Staged K-step 64 (2 WMMA sub-steps of 32), double-buffered LDS via
// global_load_async_to_lds_b128 (ASYNCcnt) + split barrier.
// ---------------------------------------------------------------------------
template <int EPI, int OUTBF>
__global__ __launch_bounds__(256) void gemm_bf16_wmma(
    const bf16_t* __restrict__ A, const bf16_t* __restrict__ Bt,
    void* __restrict__ Cout, const float* __restrict__ bias,
    int M, int N, int KP, int ldc)
{
    __shared__ __align__(16) bf16_t Alds[2][128 * LPITCH];   // [m][k0..63]
    __shared__ __align__(16) bf16_t Blds[2][128 * LPITCH];   // [n][k0..63]

    const int tid   = threadIdx.x;
    const int lane  = tid & 31;
    const int wave  = tid >> 5;
    const int waveM = wave >> 2;      // 0..1
    const int waveN = wave & 3;       // 0..3
    const int l16   = lane & 15;
    const int lhalf = lane >> 4;      // 0 or 1
    const int mBlock = blockIdx.y * 128;
    const int nBlock = blockIdx.x * 128;

    v8f acc[4][2] = {};

    // stage one 128x64 A tile + 128x64 B tile into LDS buffer `buf`
    auto stage = [&](int buf, int kb) {
#pragma unroll
        for (int it = 0; it < 4; ++it) {
            int o   = it * 256 + tid;     // 1024 octets of 8 bf16 per tile
            int row = o >> 3;             // 0..127
            int oct = o & 7;              // 0..7
            const bf16_t* ga = A  + (size_t)(mBlock + row) * KP + kb + oct * 8;
            const bf16_t* gb = Bt + (size_t)(nBlock + row) * KP + kb + oct * 8;
#if USE_ASYNC
            unsigned la = (unsigned)(uintptr_t)&Alds[buf][row * LPITCH + oct * 8];
            unsigned lb = (unsigned)(uintptr_t)&Blds[buf][row * LPITCH + oct * 8];
            asm volatile("global_load_async_to_lds_b128 %0, %1, off"
                         :: "v"(la), "v"(ga) : "memory");
            asm volatile("global_load_async_to_lds_b128 %0, %1, off"
                         :: "v"(lb), "v"(gb) : "memory");
#else
            *(u32x4*)&Alds[buf][row * LPITCH + oct * 8] = *(const u32x4*)ga;
            *(u32x4*)&Blds[buf][row * LPITCH + oct * 8] = *(const u32x4*)gb;
#endif
        }
    };

    stage(0, 0);
    wait_async0();
    __syncthreads();

    int buf = 0;
    for (int kb = 0; kb < KP; kb += 64) {
        if (kb + 64 < KP) stage(buf ^ 1, kb + 64);

#pragma unroll
        for (int ks = 0; ks < 2; ++ks) {          // two 32-wide WMMA sub-steps
            const int kc = ks * 32;
            Frag af[4], bfr[2];
#pragma unroll
            for (int mi = 0; mi < 4; ++mi) {
                const bf16_t* p = &Alds[buf][(waveM * 64 + mi * 16 + l16) * LPITCH + kc];
                af[mi].q[0] = *(const u32x4*)(p + lhalf * 8);        // K 0-7 / 8-15
                af[mi].q[1] = *(const u32x4*)(p + 16 + lhalf * 8);   // K 16-23 / 24-31
            }
#pragma unroll
            for (int ni = 0; ni < 2; ++ni) {
                const bf16_t* p = &Blds[buf][(waveN * 32 + ni * 16 + l16) * LPITCH + kc];
                bfr[ni].q[0] = *(const u32x4*)(p + lhalf * 8);       // K 0-7 / 8-15
                bfr[ni].q[1] = *(const u32x4*)(p + 16 + lhalf * 8);
            }
#pragma unroll
            for (int mi = 0; mi < 4; ++mi)
#pragma unroll
                for (int ni = 0; ni < 2; ++ni)
                    acc[mi][ni] = __builtin_amdgcn_wmma_f32_16x16x32_bf16(
                        false, af[mi].v, false, bfr[ni].v,
                        (short)0, acc[mi][ni], false, false);
        }

        wait_async0();
        __syncthreads();
        buf ^= 1;
    }

    // epilogue: 16x16 f32 D layout: n = lane&15, m = (lane>>4)*8 + g
    float*  Cf = (float*)Cout;
    bf16_t* Cb = (bf16_t*)Cout;
#pragma unroll
    for (int mi = 0; mi < 4; ++mi) {
#pragma unroll
        for (int ni = 0; ni < 2; ++ni) {
            int n = nBlock + waveN * 32 + ni * 16 + l16;
            if (n >= N) continue;
            float bv = 0.0f;
            if (EPI != EPI_NONE && bias) bv = bias[n];
#pragma unroll
            for (int g = 0; g < 8; ++g) {
                int m = mBlock + waveM * 64 + mi * 16 + lhalf * 8 + g;
                float v = acc[mi][ni][g];
                if (EPI == EPI_BIAS)               v += bv;
                else if (EPI == EPI_BIAS_RELU)     v = fmaxf(v + bv, 0.0f);
                else if (EPI == EPI_BIAS_SOFTPLUS) {
                    v += bv;
                    v = (v > 20.0f) ? v : log1pf(__expf(v));
                }
                if (OUTBF) Cb[(size_t)m * ldc + n] = (bf16_t)v;
                else       Cf[(size_t)m * ldc + n] = v;
            }
        }
    }
}

// ---------------------------------------------------------------------------
// Packing kernels
// ---------------------------------------------------------------------------
// f32 [rows x srcLd] cols [0,Ksrc) -> bf16 [rows x KP], zero-padded
__global__ __launch_bounds__(256) void pack_rows_bf16(
    const float* __restrict__ src, bf16_t* __restrict__ dst,
    int srcLd, int Ksrc, int KP)
{
    int k = blockIdx.x * 256 + threadIdx.x;
    int m = blockIdx.y;
    if (k >= KP) return;
    float v = (k < Ksrc) ? src[(size_t)m * srcLd + k] : 0.0f;
    dst[(size_t)m * KP + k] = (bf16_t)v;
}

// f32 [K x N] -> bf16 [Npad x KP] transposed, zero-padded
__global__ __launch_bounds__(256) void packT_bf16(
    const float* __restrict__ src, bf16_t* __restrict__ dst,
    int K, int N, int KP)
{
    int k = blockIdx.x * 256 + threadIdx.x;
    int n = blockIdx.y;
    if (k >= KP) return;
    float v = (k < K && n < N) ? src[(size_t)k * N + n] : 0.0f;
    dst[(size_t)n * KP + k] = (bf16_t)v;
}

// ---------------------------------------------------------------------------
// Depthwise causal conv (K=4) + SiLU over bf16 xz; bf16 out (f32 accumulate)
// ---------------------------------------------------------------------------
__global__ __launch_bounds__(256) void conv_silu_kernel(
    const bf16_t* __restrict__ xzb, const float* __restrict__ cw,
    const float* __restrict__ cb, bf16_t* __restrict__ xcbf)
{
    int d   = (blockIdx.x % 6) * 256 + threadIdx.x;   // 0..1535
    int row = blockIdx.x / 6;                         // 0..16383
    int b = row >> 12, l = row & 4095;
    float acc = cb[d];
#pragma unroll
    for (int j = 0; j < D_CONV; ++j) {
        int lp = l - (D_CONV - 1) + j;
        if (lp >= 0)
            acc += cw[d * D_CONV + j] * (float)xzb[(size_t)((b << 12) + lp) * 3072 + d];
    }
    float s = 1.0f / (1.0f + __expf(-acc));
    xcbf[(size_t)row * D_INNER + d] = (bf16_t)(acc * s);
}

// ---------------------------------------------------------------------------
// Chunked selective scan (3-pass); dt/u/z in bf16, B/C (x_dbl) in f32,
// all arithmetic and carried state in f32.
// ---------------------------------------------------------------------------
__global__ __launch_bounds__(256) void scan_pass1(
    const bf16_t* __restrict__ dt, const bf16_t* __restrict__ u,
    const float* __restrict__ xdbl, const float* __restrict__ A_log,
    float* __restrict__ P, float* __restrict__ Q)
{
    int d     = (blockIdx.x % 6) * 256 + threadIdx.x;
    int chunk = (blockIdx.x / 6) % NCHUNK;
    int b     = blockIdx.x / (6 * NCHUNK);
    float As[D_STATE], Pr[D_STATE], Qr[D_STATE];
#pragma unroll
    for (int s = 0; s < D_STATE; ++s) {
        As[s] = -__expf(A_log[d * D_STATE + s]);
        Pr[s] = 1.0f; Qr[s] = 0.0f;
    }
    int l0 = chunk * CHUNK;
    for (int l = l0; l < l0 + CHUNK; ++l) {
        int row = b * SEQ + l;
        float dtv = (float)dt[(size_t)row * D_INNER + d];
        float uv  = (float)u [(size_t)row * D_INNER + d];
        float du  = dtv * uv;
        const float* bc = &xdbl[(size_t)row * 80 + DT_RANK];
#pragma unroll
        for (int s = 0; s < D_STATE; ++s) {
            float a = __expf(dtv * As[s]);
            Qr[s] = a * Qr[s] + du * bc[s];
            Pr[s] *= a;
        }
    }
    size_t base = (size_t)((b * NCHUNK + chunk) * D_STATE) * D_INNER + d;
#pragma unroll
    for (int s = 0; s < D_STATE; ++s) {
        P[base + (size_t)s * D_INNER] = Pr[s];
        Q[base + (size_t)s * D_INNER] = Qr[s];
    }
}

__global__ __launch_bounds__(256) void scan_pass2(
    const float* __restrict__ P, const float* __restrict__ Q,
    float* __restrict__ hinit)
{
    int d = (blockIdx.x % 6) * 256 + threadIdx.x;
    int s = (blockIdx.x / 6) % D_STATE;
    int b = blockIdx.x / (6 * D_STATE);
    float h = 0.0f;
    for (int c = 0; c < NCHUNK; ++c) {
        size_t idx = (size_t)((b * NCHUNK + c) * D_STATE + s) * D_INNER + d;
        hinit[idx] = h;
        h = P[idx] * h + Q[idx];
    }
}

__global__ __launch_bounds__(256) void scan_pass3(
    const bf16_t* __restrict__ dt, const bf16_t* __restrict__ u,
    const float* __restrict__ xdbl, const bf16_t* __restrict__ xzb,
    const float* __restrict__ A_log, const float* __restrict__ Dp,
    const float* __restrict__ hinit, bf16_t* __restrict__ ybbf)
{
    int d     = (blockIdx.x % 6) * 256 + threadIdx.x;
    int chunk = (blockIdx.x / 6) % NCHUNK;
    int b     = blockIdx.x / (6 * NCHUNK);
    float As[D_STATE], hs[D_STATE];
#pragma unroll
    for (int s = 0; s < D_STATE; ++s) {
        As[s] = -__expf(A_log[d * D_STATE + s]);
        hs[s] = hinit[(size_t)((b * NCHUNK + chunk) * D_STATE + s) * D_INNER + d];
    }
    float Dd = Dp[d];
    int l0 = chunk * CHUNK;
    for (int l = l0; l < l0 + CHUNK; ++l) {
        int row = b * SEQ + l;
        float dtv = (float)dt[(size_t)row * D_INNER + d];
        float uv  = (float)u [(size_t)row * D_INNER + d];
        float du  = dtv * uv;
        const float* bc = &xdbl[(size_t)row * 80 + DT_RANK];
        const float* cc = bc + D_STATE;
        float y = 0.0f;
#pragma unroll
        for (int s = 0; s < D_STATE; ++s) {
            float a = __expf(dtv * As[s]);
            hs[s] = a * hs[s] + du * bc[s];
            y += hs[s] * cc[s];
        }
        float zv  = (float)xzb[(size_t)row * 3072 + D_INNER + d];
        float sig = 1.0f / (1.0f + __expf(-zv));
        ybbf[(size_t)row * D_INNER + d] = (bf16_t)((y + Dd * uv) * (zv * sig));
    }
}

// ---------------------------------------------------------------------------
extern "C" void kernel_launch(void* const* d_in, const int* in_sizes, int n_in,
                              void* d_out, int out_size, void* d_ws, size_t ws_size,
                              hipStream_t stream)
{
    const float* x          = (const float*)d_in[0];
    const float* in_proj_w  = (const float*)d_in[2];
    const float* conv_w     = (const float*)d_in[3];
    const float* conv_b     = (const float*)d_in[4];
    const float* x_proj_w   = (const float*)d_in[5];
    const float* dt_proj_w  = (const float*)d_in[6];
    const float* dt_proj_b  = (const float*)d_in[7];
    const float* A_log      = (const float*)d_in[8];
    const float* Dparm      = (const float*)d_in[9];
    const float* out_proj_w = (const float*)d_in[10];
    const float* head_w1    = (const float*)d_in[11];
    const float* head_b1    = (const float*)d_in[12];
    const float* head_w2    = (const float*)d_in[13];
    const float* head_b2    = (const float*)d_in[14];
    float* out = (float*)d_out;

    // ---- workspace layout ----
    char* wp = (char*)d_ws;
    auto allocF = [&](size_t n) { float*  p = (float*)wp;  wp += n * 4; return p; };
    auto allocB = [&](size_t n) { bf16_t* p = (bf16_t*)wp; wp += ((n * 2 + 15) & ~(size_t)15); return p; };

    float*  xdbl  = allocF((size_t)MROWS * 80);
    float*  Pw    = allocF((size_t)B_SZ * NCHUNK * D_STATE * D_INNER);
    float*  Qw    = allocF((size_t)B_SZ * NCHUNK * D_STATE * D_INNER);
    float*  hinit = allocF((size_t)B_SZ * NCHUNK * D_STATE * D_INNER);
    bf16_t* xzb    = allocB((size_t)MROWS * 3072);       // xz in bf16
    bf16_t* xcbf   = allocB((size_t)MROWS * D_INNER);    // conv output
    bf16_t* dtvb   = allocB((size_t)MROWS * D_INNER);    // softplus(dt)
    bf16_t* hbf    = allocB((size_t)MROWS * D_MODEL);
    bf16_t* xdblbf = allocB((size_t)MROWS * 64);
    bf16_t* ybbf   = allocB((size_t)MROWS * D_INNER);
    bf16_t* h1bf   = allocB((size_t)MROWS * D_MODEL);
    bf16_t* wInT   = allocB((size_t)3072 * D_MODEL);     // [3072 x 768]
    bf16_t* wXT    = allocB((size_t)128 * D_INNER);      // [128 x 1536]
    bf16_t* wDtT   = allocB((size_t)D_INNER * 64);       // [1536 x 64]
    bf16_t* wOutT  = allocB((size_t)D_MODEL * D_INNER);  // [768 x 1536]
    bf16_t* wH1T   = allocB((size_t)D_MODEL * D_MODEL);
    bf16_t* wH2T   = allocB((size_t)D_MODEL * D_MODEL);

    dim3 blk(256);
    const int MB = MROWS / 128;   // 128 M-blocks

    // pack input activations once: hbf = bf16(x), KP = 768
    pack_rows_bf16<<<dim3(3, MROWS), blk, 0, stream>>>(x, hbf, D_MODEL, D_MODEL, D_MODEL);

    for (int i = 0; i < LAYERS; ++i) {
        // ---- weight repacks (transposed, padded, bf16) ----
        packT_bf16<<<dim3(3, 3072), blk, 0, stream>>>(
            in_proj_w + (size_t)i * D_MODEL * 3072, wInT, D_MODEL, 3072, D_MODEL);
        packT_bf16<<<dim3(6, 128), blk, 0, stream>>>(
            x_proj_w + (size_t)i * D_INNER * 80, wXT, D_INNER, 80, D_INNER);
        packT_bf16<<<dim3(1, D_INNER), blk, 0, stream>>>(
            dt_proj_w + (size_t)i * DT_RANK * D_INNER, wDtT, DT_RANK, D_INNER, 64);
        packT_bf16<<<dim3(6, D_MODEL), blk, 0, stream>>>(
            out_proj_w + (size_t)i * D_INNER * D_MODEL, wOutT, D_INNER, D_MODEL, D_INNER);

        // xz = h @ in_proj_w[i]   (16384 x 3072, K=768) -> bf16
        gemm_bf16_wmma<EPI_NONE, 1><<<dim3(3072 / 128, MB), blk, 0, stream>>>(
            hbf, wInT, xzb, nullptr, MROWS, 3072, D_MODEL, 3072);
        // xc = silu(causal depthwise conv(xi) + cb) -> bf16
        conv_silu_kernel<<<dim3(MROWS * 6), blk, 0, stream>>>(
            xzb, conv_w + (size_t)i * D_INNER * D_CONV, conv_b + (size_t)i * D_INNER, xcbf);
        // x_dbl = xc @ x_proj_w[i]   (N=80 padded to 128, K=1536) -> f32
        gemm_bf16_wmma<EPI_NONE, 0><<<dim3(1, MB), blk, 0, stream>>>(
            xcbf, wXT, xdbl, nullptr, MROWS, 80, D_INNER, 80);
        // pack dt-rank slice: bf16 [M x 64], zero-padded from 48
        pack_rows_bf16<<<dim3(1, MROWS), blk, 0, stream>>>(xdbl, xdblbf, 80, DT_RANK, 64);
        // dt = softplus(x_dbl[:, :48] @ dt_proj_w[i] + dt_b) -> bf16
        gemm_bf16_wmma<EPI_BIAS_SOFTPLUS, 1><<<dim3(D_INNER / 128, MB), blk, 0, stream>>>(
            xdblbf, wDtT, dtvb, dt_proj_b + (size_t)i * D_INNER, MROWS, D_INNER, 64, D_INNER);
        // selective scan (chunked 3-pass), fused D*u and silu(z) gate -> bf16 y
        scan_pass1<<<dim3(6 * NCHUNK * B_SZ), blk, 0, stream>>>(
            dtvb, xcbf, xdbl, A_log + (size_t)i * D_INNER * D_STATE, Pw, Qw);
        scan_pass2<<<dim3(6 * D_STATE * B_SZ), blk, 0, stream>>>(Pw, Qw, hinit);
        scan_pass3<<<dim3(6 * NCHUNK * B_SZ), blk, 0, stream>>>(
            dtvb, xcbf, xdbl, xzb, A_log + (size_t)i * D_INNER * D_STATE,
            Dparm + (size_t)i * D_INNER, hinit, ybbf);
        // h_next = y @ out_proj_w[i] -> bf16 (feeds next in_proj / head)
        gemm_bf16_wmma<EPI_NONE, 1><<<dim3(D_MODEL / 128, MB), blk, 0, stream>>>(
            ybbf, wOutT, hbf, nullptr, MROWS, D_MODEL, D_INNER, D_MODEL);
    }

    // head
    packT_bf16<<<dim3(3, D_MODEL), blk, 0, stream>>>(head_w1, wH1T, D_MODEL, D_MODEL, D_MODEL);
    packT_bf16<<<dim3(3, D_MODEL), blk, 0, stream>>>(head_w2, wH2T, D_MODEL, D_MODEL, D_MODEL);
    gemm_bf16_wmma<EPI_BIAS_RELU, 1><<<dim3(D_MODEL / 128, MB), blk, 0, stream>>>(
        hbf, wH1T, h1bf, head_b1, MROWS, D_MODEL, D_MODEL, D_MODEL);
    gemm_bf16_wmma<EPI_BIAS, 0><<<dim3(D_MODEL / 128, MB), blk, 0, stream>>>(
        h1bf, wH2T, out, head_b2, MROWS, D_MODEL, D_MODEL, D_MODEL);
}
